// LocalAttention_1400159338531
// MI455X (gfx1250) — matrix operand order repaired and checked by reading, not compiled, not run on hardware
//
#include <hip/hip_runtime.h>
#include <hip/hip_bf16.h>
#include <math.h>

#define NTOK   8192
#define DMODEL 512
#define KROWPAD 520   // 512 elements + 8 (TDM pad: 4 dwords per 256 dwords) -> bank-decorrelated rows

typedef __bf16 bf16_t;
typedef __attribute__((ext_vector_type(16))) __bf16 v16bf;
typedef __attribute__((ext_vector_type(8)))  __bf16 v8bf;
typedef __attribute__((ext_vector_type(8)))  float  v8f;
typedef __attribute__((ext_vector_type(4)))  unsigned int u32x4;
typedef __attribute__((ext_vector_type(8)))  int i32x8;
typedef __attribute__((ext_vector_type(4)))  int i32x4;

#if __has_builtin(__builtin_amdgcn_tensor_load_to_lds)
#define LA_HAVE_TDM 1
#else
#define LA_HAVE_TDM 0
#endif

// ---------------------------------------------------------------------------
// WMMA fragment loaders (CDNA5 16-bit layouts, CDNA5 ISA §7.12.2)
// ---------------------------------------------------------------------------
static __device__ __forceinline__ v16bf load_frag_a_bf16(const bf16_t* __restrict__ A,
                                                         int lda, int lane) {
  const int m  = lane & 15;
  const int kb = (lane & 16) ? 8 : 0;
  const bf16_t* p = A + (size_t)m * lda + kb;
  v8bf c0 = *(const v8bf*)(p);
  v8bf c1 = *(const v8bf*)(p + 16);
  v16bf f;
#pragma unroll
  for (int i = 0; i < 8; ++i) { f[i] = c0[i]; f[i + 8] = c1[i]; }
  return f;
}

// B source stored "N-major": src[n*ldb + k] == B(k, n).  Works for global or LDS.
static __device__ __forceinline__ v16bf load_frag_b_bf16(const bf16_t* B,
                                                         int ldb, int lane) {
  const int n  = lane & 15;
  const int kb = (lane & 16) ? 8 : 0;
  const bf16_t* p = B + (size_t)n * ldb + kb;
  v8bf c0 = *(const v8bf*)(p);
  v8bf c1 = *(const v8bf*)(p + 16);
  v16bf f;
#pragma unroll
  for (int i = 0; i < 8; ++i) { f[i] = c0[i]; f[i + 8] = c1[i]; }
  return f;
}

static __device__ __forceinline__ v8f wmma_bf16(v16bf a, v16bf b, v8f c) {
  return __builtin_amdgcn_wmma_f32_16x16x32_bf16(false, a, false, b,
                                                 (short)0, c, false, false);
}

static __device__ __forceinline__ void atomicMaxF(float* addr, float val) {
  unsigned int* a = (unsigned int*)addr;
  unsigned int old = *a;
  while (__uint_as_float(old) < val) {
    unsigned int assumed = old;
    old = atomicCAS(a, assumed, __float_as_uint(val));
    if (old == assumed) break;
  }
}

// ---------------------------------------------------------------------------
// TDM: DMA a [rows x 512] bf16 panel (row stride DMODEL) into LDS, padding
// each 256-dword row with 4 dwords so LDS row stride = KROWPAD elements.
// Descriptor packing per CDNA5 ISA.
// ---------------------------------------------------------------------------
#if LA_HAVE_TDM
static __device__ __forceinline__ void tdm_load_panel(unsigned lds_addr,
                                                      unsigned long long gaddr,
                                                      int rows) {
  u32x4 g0;
  g0[0] = 1u;                                            // count=1, user descriptor
  g0[1] = lds_addr;                                      // lds_addr [63:32]
  g0[2] = (unsigned)(gaddr & 0xFFFFFFFFull);             // global_addr [95:64]
  g0[3] = (unsigned)((gaddr >> 32) & 0x01FFFFFFull)      // global_addr [120:96]
        | (2u << 30);                                    // type=2 ("image")
  i32x8 g1;
  g1[0] = (1 << 16)                                      // data_size = 2 bytes
        | (1 << 20)                                      // pad_enable
        | (7 << 22)                                      // pad_interval = 256 dwords
        | (3 << 25);                                     // pad_amount   = 4 dwords
  g1[1] = (DMODEL & 0xFFFF) << 16;                       // tensor_dim0[15:0]
  g1[2] = ((DMODEL >> 16) & 0xFFFF)                      // tensor_dim0[31:16]
        | ((NTOK & 0xFFFF) << 16);                       // tensor_dim1[15:0]
  g1[3] = ((NTOK >> 16) & 0xFFFF)                        // tensor_dim1[31:16]
        | ((DMODEL & 0xFFFF) << 16);                     // tile_dim0 = 512 elems
  g1[4] = rows & 0xFFFF;                                 // tile_dim1 = rows, tile_dim2 = 0
  g1[5] = DMODEL;                                        // tensor_dim0_stride[31:0]
  g1[6] = 0;
  g1[7] = 0;
  i32x4 gz = {0, 0, 0, 0};
#if defined(__clang_major__) && (__clang_major__ >= 23)
  i32x8 gz8 = {0, 0, 0, 0, 0, 0, 0, 0};
  __builtin_amdgcn_tensor_load_to_lds(g0, g1, gz, gz, gz8, 0);
#else
  __builtin_amdgcn_tensor_load_to_lds(g0, g1, gz, gz, 0);
#endif
}
#endif

// Fallback cooperative copy (also used if TDM builtin is unavailable)
static __device__ __forceinline__ void copy_panel_lds(bf16_t* dst,
                                                      const bf16_t* __restrict__ src,
                                                      int tid) {
  for (int idx = tid; idx < 128 * (DMODEL / 8); idx += 256) {
    const int r = idx >> 6;        // DMODEL/8 == 64 chunks per row
    const int c = (idx & 63) * 8;
    *(v8bf*)(dst + r * KROWPAD + c) = *(const v8bf*)(src + (size_t)r * DMODEL + c);
  }
}

// ---------------------------------------------------------------------------
// Kernel 0: f32 -> bf16 conversion
// ---------------------------------------------------------------------------
__global__ void la_cvt_bf16_kernel(const float* __restrict__ s,
                                   bf16_t* __restrict__ d, int n) {
  int i = blockIdx.x * blockDim.x + threadIdx.x;
  if (i < n) d[i] = (__bf16)s[i];
}

// ---------------------------------------------------------------------------
// Kernel 1: Q/K/V = x @ W^T  (bf16, f32 acc).  grid.z selects the weight.
// ---------------------------------------------------------------------------
__global__ __launch_bounds__(256)
void la_qkv_gemm_kernel(const bf16_t* __restrict__ xb,
                        const bf16_t* __restrict__ wb_all,
                        bf16_t* __restrict__ qb, bf16_t* __restrict__ kb,
                        bf16_t* __restrict__ vb, bf16_t* __restrict__ vt) {
  const int w = threadIdx.x >> 5, lane = threadIdx.x & 31;
  const int wr = w >> 1, wc = w & 1;
  const int r0 = blockIdx.x * 128 + wr * 32;
  const int c0 = blockIdx.y * 64 + wc * 32;
  const int z  = blockIdx.z;
  const bf16_t* W = wb_all + (size_t)z * DMODEL * DMODEL;
  bf16_t* dst = (z == 0) ? qb : (z == 1) ? kb : vb;

  v8f a00 = {}, a01 = {}, a10 = {}, a11 = {};
  for (int k = 0; k < DMODEL; k += 32) {
    __builtin_prefetch(xb + (size_t)r0 * DMODEL + k + 32, 0, 3);
    __builtin_prefetch(W + (size_t)c0 * DMODEL + k + 32, 0, 3);
    v16bf fa0 = load_frag_a_bf16(xb + (size_t)r0 * DMODEL + k, DMODEL, lane);
    v16bf fa1 = load_frag_a_bf16(xb + (size_t)(r0 + 16) * DMODEL + k, DMODEL, lane);
    v16bf fb0 = load_frag_b_bf16(W + (size_t)c0 * DMODEL + k, DMODEL, lane);
    v16bf fb1 = load_frag_b_bf16(W + (size_t)(c0 + 16) * DMODEL + k, DMODEL, lane);
    a00 = wmma_bf16(fa0, fb0, a00);
    a01 = wmma_bf16(fa0, fb1, a01);
    a10 = wmma_bf16(fa1, fb0, a10);
    a11 = wmma_bf16(fa1, fb1, a11);
  }

  const int nloc = lane & 15;
  const int mhi  = (lane & 16) ? 8 : 0;
#pragma unroll
  for (int ti = 0; ti < 2; ++ti) {
#pragma unroll
    for (int tj = 0; tj < 2; ++tj) {
      v8f acc = (ti == 0) ? ((tj == 0) ? a00 : a01) : ((tj == 0) ? a10 : a11);
#pragma unroll
      for (int i = 0; i < 8; ++i) {
        const int m = r0 + 16 * ti + mhi + i;
        const int n = c0 + 16 * tj + nloc;
        const bf16_t v = (__bf16)acc[i];
        dst[(size_t)m * DMODEL + n] = v;
        if (z == 2) vt[(size_t)n * NTOK + m] = v;
      }
    }
  }
}

// ---------------------------------------------------------------------------
// Kernel 2: masked scaled scores + stable softmax.
// One WG owns 64 rows.  K panels (128 cols x 512 k, bf16) are double-buffered
// in LDS, filled by TDM (tensor_load_to_lds) with row padding, synchronized
// with s_wait_tensorcnt + barrier.  B-fragments come from LDS; A from global.
// Pass C also emits the bf16 attn copy used by the attn@V GEMM.
// ---------------------------------------------------------------------------
__global__ __launch_bounds__(256)
void la_scores_softmax_kernel(const bf16_t* __restrict__ qb,
                              const bf16_t* __restrict__ kb,
                              const unsigned char* __restrict__ mask,
                              float* __restrict__ attn,
                              bf16_t* __restrict__ abf) {
  __shared__ __align__(16) bf16_t ksh[2][128 * KROWPAD];  // 2 x 133120 B
  __shared__ float rowmax[64];
  __shared__ float rowsum[64];

  const int tid  = threadIdx.x;
  const int w    = tid >> 5, lane = tid & 31;
  const int wr   = w >> 2, wc = w & 3;      // 2 x 4 waves
  const int r0   = blockIdx.x * 64;
  const int rr   = r0 + wr * 32;
  const float scale = 0.044194173824159216f;  // 1/sqrt(512)

  if (tid < 64) { rowmax[tid] = -3.0e38f; rowsum[tid] = 0.0f; }

  // ---- preload K panel 0 --------------------------------------------------
#if LA_HAVE_TDM
  if (tid < 32) {
    tdm_load_panel((unsigned)(uintptr_t)(void*)&ksh[0][0],
                   (unsigned long long)(uintptr_t)kb, 128);
    __builtin_amdgcn_s_wait_tensorcnt(0);
  }
#else
  copy_panel_lds(&ksh[0][0], kb, tid);
#endif
  __syncthreads();

  const int nloc = lane & 15;
  const int mhi  = (lane & 16) ? 8 : 0;

  // ---- pass A: raw masked scaled scores + row max -------------------------
  for (int ct = 0; ct < NTOK; ct += 128) {
    const int cur = (ct >> 7) & 1;
#if LA_HAVE_TDM
    if (tid < 32 && (ct + 128) < NTOK) {
      tdm_load_panel((unsigned)(uintptr_t)(void*)&ksh[cur ^ 1][0],
                     (unsigned long long)(uintptr_t)(kb + (size_t)(ct + 128) * DMODEL),
                     128);
    }
#else
    if ((ct + 128) < NTOK) {
      copy_panel_lds(&ksh[cur ^ 1][0], kb + (size_t)(ct + 128) * DMODEL, tid);
    }
#endif
    const int c0 = ct + wc * 32;
    const bf16_t* kpan = &ksh[cur][(size_t)(wc * 32) * KROWPAD];

    v8f a00 = {}, a01 = {}, a10 = {}, a11 = {};
    for (int k = 0; k < DMODEL; k += 32) {
      __builtin_prefetch(qb + (size_t)rr * DMODEL + k + 32, 0, 3);
      v16bf fa0 = load_frag_a_bf16(qb + (size_t)rr * DMODEL + k, DMODEL, lane);
      v16bf fa1 = load_frag_a_bf16(qb + (size_t)(rr + 16) * DMODEL + k, DMODEL, lane);
      v16bf fb0 = load_frag_b_bf16(kpan + k, KROWPAD, lane);
      v16bf fb1 = load_frag_b_bf16(kpan + (size_t)16 * KROWPAD + k, KROWPAD, lane);
      a00 = wmma_bf16(fa0, fb0, a00);
      a01 = wmma_bf16(fa0, fb1, a01);
      a10 = wmma_bf16(fa1, fb0, a10);
      a11 = wmma_bf16(fa1, fb1, a11);
    }
#pragma unroll
    for (int ti = 0; ti < 2; ++ti) {
#pragma unroll
      for (int tj = 0; tj < 2; ++tj) {
        v8f acc = (ti == 0) ? ((tj == 0) ? a00 : a01) : ((tj == 0) ? a10 : a11);
#pragma unroll
        for (int i = 0; i < 8; ++i) {
          const int m = rr + 16 * ti + mhi + i;
          const int n = c0 + 16 * tj + nloc;
          float s = acc[i] * scale;
          if (mask[(size_t)m * NTOK + n] != 0) s = -1.0e9f;
          attn[(size_t)m * NTOK + n] = s;
          atomicMaxF(&rowmax[m - r0], s);
        }
      }
    }
#if LA_HAVE_TDM
    if (tid < 32) __builtin_amdgcn_s_wait_tensorcnt(0);
#endif
    __syncthreads();
  }
  __threadfence();
  __syncthreads();

  // ---- pass B: accumulate row sums of exp(s - rowmax) (no store) ----------
  const int row = tid >> 2;   // 0..63
  const int qid = tid & 3;    // 4 threads per row
  float* rowp = attn + (size_t)(r0 + row) * NTOK;
  bf16_t* brow = abf + (size_t)(r0 + row) * NTOK;
  const float rmax = rowmax[row];
  float lsum = 0.0f;
  for (int j = 0; j < NTOK / 4; ++j) {
    lsum += __expf(rowp[qid + 4 * j] - rmax);
  }
  atomicAdd(&rowsum[row], lsum);
  __syncthreads();

  // ---- pass C: write normalized attn (f32) + bf16 copy --------------------
  const float inv = 1.0f / rowsum[row];
  for (int j = 0; j < NTOK / 4; ++j) {
    const int col = qid + 4 * j;
    const float e = __expf(rowp[col] - rmax) * inv;
    rowp[col] = e;
    brow[col] = (__bf16)e;
  }
}

// ---------------------------------------------------------------------------
// Kernel 3: cntx = attn @ V.  attn read as bf16 (abf), B from V^T (bf16).
// WG tile 64x256 (grid.y = 2 -> attn streamed only twice), wave tile 32x64.
// ---------------------------------------------------------------------------
__global__ __launch_bounds__(256)
void la_cntx_gemm_kernel(const bf16_t* __restrict__ abf,
                         const bf16_t* __restrict__ vt,
                         float* __restrict__ out) {
  const int w = threadIdx.x >> 5, lane = threadIdx.x & 31;
  const int wr = w >> 2, wc = w & 3;          // 2 x 4 waves
  const int r0 = blockIdx.x * 64 + wr * 32;
  const int c0 = blockIdx.y * 256 + wc * 64;

  v8f vzero = {};
  v8f acc[2][4];
#pragma unroll
  for (int ti = 0; ti < 2; ++ti)
#pragma unroll
    for (int tj = 0; tj < 4; ++tj) acc[ti][tj] = vzero;

  for (int k = 0; k < NTOK; k += 32) {
    __builtin_prefetch(abf + (size_t)r0 * NTOK + k + 32, 0, 3);
    v16bf fa0 = load_frag_a_bf16(abf + (size_t)r0 * NTOK + k, NTOK, lane);
    v16bf fa1 = load_frag_a_bf16(abf + (size_t)(r0 + 16) * NTOK + k, NTOK, lane);
#pragma unroll
    for (int tj = 0; tj < 4; ++tj) {
      v16bf fb = load_frag_b_bf16(vt + (size_t)(c0 + 16 * tj) * NTOK + k, NTOK, lane);
      acc[0][tj] = wmma_bf16(fa0, fb, acc[0][tj]);
      acc[1][tj] = wmma_bf16(fa1, fb, acc[1][tj]);
    }
  }

  const int nloc = lane & 15;
  const int mhi  = (lane & 16) ? 8 : 0;
#pragma unroll
  for (int ti = 0; ti < 2; ++ti) {
#pragma unroll
    for (int tj = 0; tj < 4; ++tj) {
#pragma unroll
      for (int i = 0; i < 8; ++i) {
        const int m = r0 + 16 * ti + mhi + i;
        const int n = c0 + 16 * tj + nloc;
        out[(size_t)m * DMODEL + n] = acc[ti][tj][i];
      }
    }
  }
}

// ---------------------------------------------------------------------------
extern "C" void kernel_launch(void* const* d_in, const int* in_sizes, int n_in,
                              void* d_out, int out_size, void* d_ws, size_t ws_size,
                              hipStream_t stream) {
  const float* x  = (const float*)d_in[0];
  const unsigned char* mask = (const unsigned char*)d_in[1];
  const float* Wq = (const float*)d_in[2];
  const float* Wk = (const float*)d_in[3];
  const float* Wv = (const float*)d_in[4];

  float* out  = (float*)d_out;
  float* cntx = out;                               // [NTOK, DMODEL]
  float* attn = out + (size_t)NTOK * DMODEL;       // [NTOK, NTOK]

  char* ws = (char*)d_ws;
  size_t off = 0;
  auto carve = [&](size_t bytes) -> char* {
    char* p = ws + off;
    off += (bytes + 255) & ~(size_t)255;
    return p;
  };
  bf16_t* xb  = (bf16_t*)carve((size_t)NTOK * DMODEL * 2);
  bf16_t* wb  = (bf16_t*)carve((size_t)3 * DMODEL * DMODEL * 2);
  bf16_t* qb  = (bf16_t*)carve((size_t)NTOK * DMODEL * 2);
  bf16_t* kbp = (bf16_t*)carve((size_t)NTOK * DMODEL * 2);
  bf16_t* vb  = (bf16_t*)carve((size_t)NTOK * DMODEL * 2);
  bf16_t* vt  = (bf16_t*)carve((size_t)NTOK * DMODEL * 2);  // V^T, [DMODEL, NTOK]
  bf16_t* abf = (bf16_t*)carve((size_t)NTOK * NTOK * 2);    // attn in bf16

  // Phase 0: downconvert operands to bf16
  {
    const int n = NTOK * DMODEL;
    la_cvt_bf16_kernel<<<(n + 255) / 256, 256, 0, stream>>>(x, xb, n);
    const int nw = DMODEL * DMODEL;
    la_cvt_bf16_kernel<<<(nw + 255) / 256, 256, 0, stream>>>(Wq, wb, nw);
    la_cvt_bf16_kernel<<<(nw + 255) / 256, 256, 0, stream>>>(Wk, wb + (size_t)DMODEL * DMODEL, nw);
    la_cvt_bf16_kernel<<<(nw + 255) / 256, 256, 0, stream>>>(Wv, wb + (size_t)2 * DMODEL * DMODEL, nw);
  }

  // Phase 1: Q/K/V projections (WMMA bf16)
  la_qkv_gemm_kernel<<<dim3(NTOK / 128, DMODEL / 64, 3), 256, 0, stream>>>(
      xb, wb, qb, kbp, vb, vt);

  // Phase 2: masked scores + softmax (WMMA bf16, TDM-staged K panels in LDS)
  la_scores_softmax_kernel<<<dim3(NTOK / 64), 256, 0, stream>>>(qb, kbp, mask, attn, abf);

  // Phase 3: cntx = attn @ V (WMMA bf16)
  la_cntx_gemm_kernel<<<dim3(NTOK / 64, DMODEL / 256), 256, 0, stream>>>(abf, vt, cntx);
}
